// GAT_Reddit_74062416052498
// MI455X (gfx1250) — matrix operand order repaired and checked by reading, not compiled, not run on hardware
//
#include <hip/hip_runtime.h>
#include <math.h>

// ---------------- problem constants (from reference) ----------------
#define N0    120000
#define NN1   12000
#define NN2   1024
#define EE1   300000
#define EE2   10240
#define F_IN  602
#define HC1   64      // H1*C1 = 8*8
#define H1H   8
#define C1C   8
#define NCLS  41
#define NEG_SLOPE 0.2f

// ---------------- WMMA types ----------------
typedef __bf16 bf16_t;
typedef __attribute__((ext_vector_type(16))) __bf16 v16bf;
typedef __attribute__((ext_vector_type(8)))  __bf16 v8bf;
typedef __attribute__((ext_vector_type(8)))  float  v8f;

static __host__ __device__ inline int cdiv_i(int a, int b) { return (a + b - 1) / b; }

__device__ __forceinline__ v16bf cat16(v8bf lo, v8bf hi) {
    return __builtin_shufflevector(lo, hi,
        0, 1, 2, 3, 4, 5, 6, 7, 8, 9, 10, 11, 12, 13, 14, 15);
}

// ---------------- generic fill ----------------
__global__ void fill_kernel(float* __restrict__ p, float v, int n) {
    int i = blockIdx.x * blockDim.x + threadIdx.x;
    if (i < n) p[i] = v;
}

// ---------------- bf16 WMMA GEMM: C[M x Nstride] = A[M x K] * B[K x N] ----------------
// A row-major (lda = K), B row-major (ldb = N), fp32 accumulate.
// Block = 256 threads = 8 waves; NPAD/16 col tiles; rowsPerBlock = (8/(NPAD/16))*16.
// Per K-chunk (KC % 32 == 0, Kpad % KC == 0):
//   - A tile staged to LDS bf16, coalesced + zero-padded
//   - B chunk staged to LDS pre-swizzled into fragment order:
//       ldsB[((k0/32)*NPAD + col)*32 + (k%32)] -> each lane reads 2x16B contiguous
// Fragment loads are software-pipelined one 32-K step ahead of the WMMA that
// consumes them, so the ds wait before each WMMA leaves the next step's loads
// in flight instead of draining the LDS pipe.
template <int KC, int NPAD>
__global__ __launch_bounds__(256)
void gemm_bf16_wmma(const float* __restrict__ A, const float* __restrict__ B,
                    float* __restrict__ C, int M, int K, int Kpad, int N, int Nstride) {
    constexpr int NTILES = NPAD / 16;
    constexpr int ROWTILES = 8 / NTILES;
    constexpr int ROWS = ROWTILES * 16;
    constexpr int KSTEPS = KC / 32;
    constexpr int BSTRIDE = NPAD * 32;   // elements per 32-K swizzled tile

    extern __shared__ bf16_t lds[];
    bf16_t* ldsA = lds;                  // ROWS * KC
    bf16_t* ldsB = lds + ROWS * KC;      // KC * NPAD (swizzled)

    const int tid  = threadIdx.x;
    const int wave = tid >> 5;
    const int lane = tid & 31;
    const int m_   = lane & 15;
    const int hi   = lane >> 4;          // 0: lanes 0-15, 1: lanes 16-31

    const int rowTile = wave / NTILES;
    const int colTile = wave % NTILES;
    const bool active = (rowTile < ROWTILES);

    const int rowBase = blockIdx.x * ROWS;
    const int row0 = rowBase + rowTile * 16;
    const int col  = colTile * 16 + m_;

    v8f acc = {};

    for (int kc = 0; kc < Kpad; kc += KC) {
        // ---- stage A tile: coalesced along K, zero-padded, fp32 -> bf16 ----
        #pragma unroll 4
        for (int idx = tid; idx < ROWS * KC; idx += 256) {
            int r = idx / KC;
            int k = idx - r * KC;
            int gk = kc + k;
            float v = (gk < K) ? A[(size_t)(rowBase + r) * (size_t)K + gk] : 0.0f;
            ldsA[idx] = (bf16_t)v;
        }
        // ---- stage B chunk, swizzled to fragment order ----
        #pragma unroll 4
        for (int d = tid; d < KC * NPAD; d += 256) {
            int q    = d & 31;
            int rest = d >> 5;
            int n    = rest % NPAD;
            int t    = rest / NPAD;
            int gk   = kc + t * 32 + q;
            float v = (gk < K && n < N) ? B[(size_t)gk * (size_t)N + n] : 0.0f;
            ldsB[d] = (bf16_t)v;
        }
        // prefetch next A chunk while we compute on this one
        if (kc + KC < K && tid < ROWS) {
            __builtin_prefetch(&A[(size_t)(rowBase + tid) * (size_t)K + kc + KC], 0, 0);
        }
        __syncthreads();

        if (active) {
            const bf16_t* arow = ldsA + (rowTile * 16 + m_) * KC + hi * 16;
            const bf16_t* bcol = ldsB + ((col << 5) + hi * 16);

            // pipeline: preload step 0
            v8bf alo = *(const v8bf*)(arow);
            v8bf ahi = *(const v8bf*)(arow + 8);
            v8bf blo = *(const v8bf*)(bcol);
            v8bf bhi = *(const v8bf*)(bcol + 8);

            #pragma unroll
            for (int s = 0; s < KSTEPS; ++s) {
                v8bf nalo = alo, nahi = ahi, nblo = blo, nbhi = bhi;
                if (s + 1 < KSTEPS) {
                    const bf16_t* an = arow + (s + 1) * 32;
                    const bf16_t* bn = bcol + (s + 1) * BSTRIDE;
                    nalo = *(const v8bf*)(an);
                    nahi = *(const v8bf*)(an + 8);
                    nblo = *(const v8bf*)(bn);
                    nbhi = *(const v8bf*)(bn + 8);
                }
                acc = __builtin_amdgcn_wmma_f32_16x16x32_bf16(
                        false, cat16(alo, ahi), false, cat16(blo, bhi),
                        (short)0, acc, false, false);
                alo = nalo; ahi = nahi; blo = nblo; bhi = nbhi;
            }
        }
        __syncthreads();
    }

    if (active) {
        #pragma unroll
        for (int r = 0; r < 8; ++r) {
            int rr = row0 + hi * 8 + r;
            C[(size_t)rr * (size_t)Nstride + col] = acc[r];
        }
    }
}

// ---------------- attention coefficients: a_s[n,h] = sum_c H[n,h*C+c]*att_s[h*C+c] ----------------
__global__ void att_coef_kernel(const float* __restrict__ Hm, int stride, int Hh, int C,
                                const float* __restrict__ att_s, const float* __restrict__ att_d,
                                float* __restrict__ a_s, float* __restrict__ a_d,
                                int Nsrc, int Ndst) {
    int idx = blockIdx.x * blockDim.x + threadIdx.x;
    if (idx >= Nsrc * Hh) return;
    int n = idx / Hh, h = idx - n * Hh;
    const float* row = Hm + (size_t)n * stride + h * C;
    float s = 0.f, d = 0.f;
    for (int c = 0; c < C; ++c) {
        float v = row[c];
        s += v * att_s[h * C + c];
        d += v * att_d[h * C + c];
    }
    a_s[idx] = s;
    if (n < Ndst) a_d[n * Hh + h] = d;
}

// ---------------- float atomic max via signed/unsigned integer ordering ----------------
__device__ __forceinline__ void atomicMaxFloat(float* addr, float val) {
    if (val >= 0.0f) atomicMax((int*)addr, __float_as_int(val));
    else             atomicMin((unsigned int*)addr, (unsigned int)__float_as_int(val));
}

// entry e in [0, E+nloop): e < E -> edge (ei[e], ei[E+e]); else self-loop node e-E.
__device__ __forceinline__ void entry_src_dst(const int* ei, int E, int e, int& src, int& dst) {
    if (e < E) { src = ei[e]; dst = ei[E + e]; }
    else       { src = dst = e - E; }
}

// pass 1: raw leaky-relu logits + segment max
__global__ void edge_max_kernel(const int* __restrict__ ei, int E, int nloop, int Hh,
                                const float* __restrict__ a_s, const float* __restrict__ a_d,
                                float* __restrict__ alpha, float* __restrict__ mx) {
    int idx = blockIdx.x * blockDim.x + threadIdx.x;
    int total = (E + nloop) * Hh;
    if (idx >= total) return;
    int e = idx / Hh, h = idx - e * Hh;
    int src, dst; entry_src_dst(ei, E, e, src, dst);
    float al = a_s[src * Hh + h] + a_d[dst * Hh + h];
    al = (al > 0.f) ? al : NEG_SLOPE * al;
    alpha[idx] = al;
    atomicMaxFloat(&mx[dst * Hh + h], al);
}

// pass 2: exp(alpha - max), segment sum
__global__ void edge_exp_kernel(const int* __restrict__ ei, int E, int nloop, int Hh,
                                float* __restrict__ alpha, const float* __restrict__ mx,
                                float* __restrict__ sm) {
    int idx = blockIdx.x * blockDim.x + threadIdx.x;
    int total = (E + nloop) * Hh;
    if (idx >= total) return;
    int e = idx / Hh, h = idx - e * Hh;
    int src, dst; entry_src_dst(ei, E, e, src, dst);
    (void)src;
    float ex = expf(alpha[idx] - mx[dst * Hh + h]);
    alpha[idx] = ex;
    atomicAdd(&sm[dst * Hh + h], ex);
}

// pass 3: out[dst, h*C + c] += Hsrc[src, h*C + c] * alpha/(sum+eps)
__global__ void edge_agg_kernel(const int* __restrict__ ei, int E, int nloop, int Hh, int C,
                                int srcStride, int dstStride,
                                const float* __restrict__ alpha, const float* __restrict__ sm,
                                const float* __restrict__ Hsrc, float* __restrict__ out) {
    int idx = blockIdx.x * blockDim.x + threadIdx.x;
    int total = (E + nloop) * Hh;
    if (idx >= total) return;
    int e = idx / Hh, h = idx - e * Hh;
    int src, dst; entry_src_dst(ei, E, e, src, dst);
    float w = alpha[idx] / (sm[dst * Hh + h] + 1e-16f);
    const float* hs = Hsrc + (size_t)src * srcStride + h * C;
    float* op = out + (size_t)dst * dstStride + h * C;
    for (int c = 0; c < C; ++c) atomicAdd(&op[c], hs[c] * w);
}

// bias + ELU (layer-1 epilogue)
__global__ void bias_elu_kernel(float* __restrict__ o, const float* __restrict__ b, int n) {
    int i = blockIdx.x * blockDim.x + threadIdx.x;
    if (i >= n) return;
    float v = o[i] + b[i & (HC1 - 1)];
    o[i] = (v > 0.f) ? v : (expf(v) - 1.0f);
}

// bias + log_softmax over 41 classes -> d_out
__global__ void logsoftmax_kernel(const float* __restrict__ o2, const float* __restrict__ b2,
                                  float* __restrict__ out) {
    int t = blockIdx.x * blockDim.x + threadIdx.x;
    if (t >= NN2) return;
    float v[NCLS];
    float m = -3.0e38f;
    for (int c = 0; c < NCLS; ++c) {
        v[c] = o2[t * NCLS + c] + b2[c];
        m = fmaxf(m, v[c]);
    }
    float s = 0.f;
    for (int c = 0; c < NCLS; ++c) s += expf(v[c] - m);
    float l = logf(s);
    for (int c = 0; c < NCLS; ++c) out[t * NCLS + c] = v[c] - m - l;
}

// ---------------- host launch ----------------
extern "C" void kernel_launch(void* const* d_in, const int* in_sizes, int n_in,
                              void* d_out, int out_size, void* d_ws, size_t ws_size,
                              hipStream_t stream) {
    const float* x   = (const float*)d_in[0];
    const int*   ei1 = (const int*)d_in[1];
    const int*   ei2 = (const int*)d_in[2];
    // d_in[3], d_in[4]: size1/size2 scalars (compile-time constants here)
    const float* W1  = (const float*)d_in[5];
    const float* as1 = (const float*)d_in[6];
    const float* ad1 = (const float*)d_in[7];
    const float* b1  = (const float*)d_in[8];
    const float* W2  = (const float*)d_in[9];
    const float* as2 = (const float*)d_in[10];
    const float* ad2 = (const float*)d_in[11];
    const float* b2  = (const float*)d_in[12];
    float* out = (float*)d_out;

    // workspace carve-up (floats); total ~51.4 MB
    float* p = (float*)d_ws;
    float* H1buf = p;                 p += (size_t)N0 * HC1;        // 120000*64
    float* aS1   = p;                 p += (size_t)N0 * H1H;        // 120000*8
    float* aD1   = p;                 p += (size_t)NN1 * H1H;       // 12000*8
    float* alp1  = p;                 p += (size_t)(EE1 + NN1) * H1H;
    float* mx1   = p;                 p += (size_t)NN1 * H1H;
    float* sm1   = p;                 p += (size_t)NN1 * H1H;
    float* o1    = p;                 p += (size_t)NN1 * HC1;       // 12000*64
    float* h2    = p;                 p += (size_t)NN1 * 48;        // padded N=48
    float* aS2   = p;                 p += (size_t)NN1;
    float* aD2   = p;                 p += (size_t)NN2;
    float* alp2  = p;                 p += (size_t)(EE2 + NN2);
    float* mx2   = p;                 p += (size_t)NN2;
    float* sm2   = p;                 p += (size_t)NN2;
    float* o2    = p;                 p += (size_t)NN2 * NCLS;

    const int TB = 256;

    // ---- init accumulators ----
    fill_kernel<<<cdiv_i(NN1 * H1H, TB), TB, 0, stream>>>(mx1, -3.0e38f, NN1 * H1H);
    fill_kernel<<<cdiv_i(NN1 * H1H, TB), TB, 0, stream>>>(sm1, 0.0f, NN1 * H1H);
    fill_kernel<<<cdiv_i(NN1 * HC1, TB), TB, 0, stream>>>(o1, 0.0f, NN1 * HC1);
    fill_kernel<<<cdiv_i(NN2, TB), TB, 0, stream>>>(mx2, -3.0e38f, NN2);
    fill_kernel<<<cdiv_i(NN2, TB), TB, 0, stream>>>(sm2, 0.0f, NN2);
    fill_kernel<<<cdiv_i(NN2 * NCLS, TB), TB, 0, stream>>>(o2, 0.0f, NN2 * NCLS);

    // ---- layer 1: H = x @ W1 (WMMA bf16, fp32 acc). M=120000 (=3750*32), K=602->640, N=64 ----
    {
        // LDS: A tile 32*320 bf16 (20480 B) + swizzled B 320*64 bf16 (40960 B) = 61440 B
        size_t shmem = (size_t)(32 * 320 + 320 * 64) * sizeof(bf16_t);
        gemm_bf16_wmma<320, 64><<<N0 / 32, TB, shmem, stream>>>(
            x, W1, H1buf, N0, F_IN, 640, HC1, HC1);
    }
    // attention coefficients
    att_coef_kernel<<<cdiv_i(N0 * H1H, TB), TB, 0, stream>>>(
        H1buf, HC1, H1H, C1C, as1, ad1, aS1, aD1, N0, NN1);
    // segmented softmax + aggregate (E1 edges + N1 self-loops, 8 heads)
    {
        int total = (EE1 + NN1) * H1H;
        edge_max_kernel<<<cdiv_i(total, TB), TB, 0, stream>>>(ei1, EE1, NN1, H1H, aS1, aD1, alp1, mx1);
        edge_exp_kernel<<<cdiv_i(total, TB), TB, 0, stream>>>(ei1, EE1, NN1, H1H, alp1, mx1, sm1);
        edge_agg_kernel<<<cdiv_i(total, TB), TB, 0, stream>>>(ei1, EE1, NN1, H1H, C1C,
                                                              HC1, HC1, alp1, sm1, H1buf, o1);
    }
    bias_elu_kernel<<<cdiv_i(NN1 * HC1, TB), TB, 0, stream>>>(o1, b1, NN1 * HC1);

    // ---- layer 2: h2 = o1 @ W2 (WMMA). M=12000 (=375*32), K=64, N=41->48 ----
    {
        // LDS: A tile 32*64 bf16 (4096 B) + swizzled B 64*48 bf16 (6144 B) = 10240 B
        size_t shmem = (size_t)(32 * 64 + 64 * 48) * sizeof(bf16_t);
        gemm_bf16_wmma<64, 48><<<NN1 / 32, TB, shmem, stream>>>(
            o1, W2, h2, NN1, HC1, 64, NCLS, 48);
    }
    att_coef_kernel<<<cdiv_i(NN1, TB), TB, 0, stream>>>(
        h2, 48, 1, NCLS, as2, ad2, aS2, aD2, NN1, NN2);
    {
        int total = (EE2 + NN2);
        edge_max_kernel<<<cdiv_i(total, TB), TB, 0, stream>>>(ei2, EE2, NN2, 1, aS2, aD2, alp2, mx2);
        edge_exp_kernel<<<cdiv_i(total, TB), TB, 0, stream>>>(ei2, EE2, NN2, 1, alp2, mx2, sm2);
        edge_agg_kernel<<<cdiv_i(total, TB), TB, 0, stream>>>(ei2, EE2, NN2, 1, NCLS,
                                                              48, NCLS, alp2, sm2, h2, o2);
    }
    logsoftmax_kernel<<<cdiv_i(NN2, TB), TB, 0, stream>>>(o2, b2, out);
}
